// CRF_23587960389772
// MI455X (gfx1250) — matrix operand order repaired and checked by reading, not compiled, not run on hardware
//
#include <hip/hip_runtime.h>

typedef __attribute__((ext_vector_type(16))) _Float16 v16h;
typedef __attribute__((ext_vector_type(2)))  _Float16 v2h;
typedef __attribute__((ext_vector_type(8)))  float    v8f;
typedef __attribute__((ext_vector_type(4)))  unsigned int v4u;
typedef __attribute__((ext_vector_type(8)))  int      v8i;
typedef __attribute__((ext_vector_type(4)))  int      v4i;

#define B_ 256
#define T_ 1024
#define K_ 64

#define LOG2E_F 1.44269504088896340736f
#define LN2_F   0.69314718055994530942f

// Raw hardware transcendentals: v_exp_f32 (2^x) and v_log_f32 (log2 x).
__device__ __forceinline__ float hw_exp2(float x) { return __builtin_amdgcn_exp2f(x); }
__device__ __forceinline__ float hw_log2(float x) { return __builtin_amdgcn_logf(x); }

#if __has_builtin(__builtin_amdgcn_tensor_load_to_lds) && __has_builtin(__builtin_amdgcn_s_wait_tensorcnt)
#define CRF_HAVE_TDM 1
#else
#define CRF_HAVE_TDM 0
#endif

// ---------------------------------------------------------------------------
// Kernel 0: E = exp(transition), split into f16 hi/lo pair (compensated WMMA)
// ---------------------------------------------------------------------------
__global__ void crf_prep_exp(const float* __restrict__ trans,
                             _Float16* __restrict__ Ehi,
                             _Float16* __restrict__ Elo) {
  int i = blockIdx.x * blockDim.x + threadIdx.x;
  if (i < K_ * K_) {
    float e = hw_exp2(trans[i] * LOG2E_F);
    _Float16 h = (_Float16)e;
    Ehi[i] = h;
    Elo[i] = (_Float16)(e - (float)h);
  }
}

// ---------------------------------------------------------------------------
// Kernel 1: per-batch emission + transition gather scores
// ---------------------------------------------------------------------------
__global__ void crf_scores(const int* __restrict__ y,
                           const float* __restrict__ em,
                           const float* __restrict__ trans,
                           float* __restrict__ scores) {
  const int b = blockIdx.x;
  const int tid = threadIdx.x;
  float s = 0.f;
  for (int t = tid; t < T_; t += 256) {
    int yt = y[b * T_ + t];
    if (yt != 0) {                       // mask[b,t]
      s += em[(b * T_ + t) * K_ + yt];   // e_score
      if (t >= 1) {
        int yp = y[b * T_ + t - 1];
        s += trans[yp * K_ + yt];        // t_score
      }
    }
  }
  #pragma unroll
  for (int off = 16; off; off >>= 1) s += __shfl_xor(s, off, 32);
  __shared__ float part[8];
  if ((tid & 31) == 0) part[tid >> 5] = s;
  __syncthreads();
  if (tid == 0) {
    float tot = 0.f;
    #pragma unroll
    for (int w = 0; w < 8; ++w) tot += part[w];
    scores[b] = tot;
  }
}

#if CRF_HAVE_TDM
// Issue a TDM load of one 16-row x 64-col f32 tile (row stride T_*K_) to LDS.
// D# packing per CDNA5 ISA sec 8.3-8.5 (2D tensor; groups 2/3 and the extra
// group of the 6-arg clang-23 builtin are zero-filled).
__device__ __forceinline__ void crf_tdm_tile(const float* gsrc, unsigned lds_off) {
  unsigned long long ga = (unsigned long long)(size_t)gsrc;
  v4u g0;
  g0.x = 1u;                                            // count=1, user mode
  g0.y = lds_off;                                       // lds_addr (bytes)
  g0.z = (unsigned)(ga & 0xffffffffu);                  // global_addr[31:0]
  g0.w = (unsigned)((ga >> 32) & 0x01ffffffu)           // global_addr[56:32]
         | (2u << 30);                                  // type=2 ("image")
  v8i g1;
  g1[0] = 0x00020000;          // workgroup_mask=0 | data_size=2 (4B)
  g1[1] = (int)(64u << 16);    // atomic_barrier_addr=0 | tensor_dim0.lo16=64
  g1[2] = (int)(16u << 16);    // tensor_dim0.hi16=0    | tensor_dim1.lo16=16
  g1[3] = (int)(64u << 16);    // tensor_dim1.hi16=0    | tile_dim0=64
  g1[4] = 16;                  // tile_dim1=16          | tile_dim2=0
  g1[5] = (int)(T_ * K_);      // tensor_dim0_stride[31:0] = 65536 elements
  g1[6] = 0;                   // stride0[47:32]=0 | tensor_dim1_stride.lo16=0
  g1[7] = 0;                   // tensor_dim1_stride[47:16]=0
  v4i gz4 = {0, 0, 0, 0};
  v8i gz8 = {0, 0, 0, 0, 0, 0, 0, 0};
  __builtin_amdgcn_tensor_load_to_lds(g0, g1, gz4, gz4, gz8, 0);
}
#endif

// ---------------------------------------------------------------------------
// Kernel 2: forward-algorithm scan as a chain of compensated f16 WMMAs.
// One block = 16 batch rows, 4 waves; wave w owns output columns 16w..16w+15.
// Emission tiles are double-buffered through LDS by the Tensor Data Mover.
// ---------------------------------------------------------------------------
__global__ __launch_bounds__(128)
void crf_scan(const int* __restrict__ y,
              const float* __restrict__ em,
              const _Float16* __restrict__ Ehi,
              const _Float16* __restrict__ Elo,
              float* __restrict__ logz) {
  __shared__ float    alpha[16][K_];
  __shared__ float    rowmax[16];
  __shared__ float    ymsk[16];
  __shared__ _Float16 ahi[16][K_];
  __shared__ _Float16 alo[16][K_];
  __shared__ float    ebuf[2][16][K_];

  const int tid  = threadIdx.x;
  const int wave = tid >> 5;
  const int lane = tid & 31;
  const int b0   = blockIdx.x * 16;
  const int row8 = tid >> 3;          // 0..15  (8 threads per row)
  const int sub8 = tid & 7;
  const int c0   = sub8 * 8;          // contiguous 8-col run per thread

  // Constant B fragments (32x16 f16 layout: N = lane%16, K = e + 16*(lane/16))
  v16h bh0, bh1, bl0, bl1;
  {
    const int n  = 16 * wave + (lane & 15);
    const int kb = (lane & 16) ? 16 : 0;
    #pragma unroll
    for (int e = 0; e < 16; ++e) {
      int k0 = kb + e;        // K in [0,32)
      int k1 = kb + e + 32;   // K in [32,64)
      bh0[e] = Ehi[k0 * K_ + n];  bl0[e] = Elo[k0 * K_ + n];
      bh1[e] = Ehi[k1 * K_ + n];  bl1[e] = Elo[k1 * K_ + n];
    }
  }

  // alpha0 = emissions[:, 0, :]
  #pragma unroll
  for (int i = 0; i < 8; ++i)
    alpha[row8][c0 + i] = em[((size_t)(b0 + row8) * T_) * K_ + c0 + i];

  // Kick off emission tile for t=1 into buffer 1.
#if CRF_HAVE_TDM
  if (wave == 0)
    crf_tdm_tile(em + ((size_t)b0 * T_ + 1) * K_,
                 (unsigned)(size_t)&ebuf[1][0][0]);
#else
  #pragma unroll
  for (int i = 0; i < 8; ++i)
    ebuf[1][row8][c0 + i] = em[((size_t)(b0 + row8) * T_ + 1) * K_ + c0 + i];
#endif

  const int jcol = 16 * wave + (lane & 15);     // C/D: N = lane%16
  const int rofs = (lane & 16) ? 8 : 0;         // C/D: M = r + 8*(lane/16)

  for (int t = 1; t < T_; ++t) {
    __syncthreads();                            // prev alpha writes / buf free
    // ---- row-wise max of alpha; 16 threads fetch the step's row masks ----
    {
      float mx = alpha[row8][c0];
      #pragma unroll
      for (int c = 1; c < 8; ++c) mx = fmaxf(mx, alpha[row8][c0 + c]);
      mx = fmaxf(mx, __shfl_xor(mx, 1, 32));
      mx = fmaxf(mx, __shfl_xor(mx, 2, 32));
      mx = fmaxf(mx, __shfl_xor(mx, 4, 32));
      if (sub8 == 0) rowmax[row8] = mx;
      if (tid < 16)
        ymsk[tid] = (y[(size_t)(b0 + tid) * T_ + t] != 0) ? 1.f : 0.f;
    }
    // ---- fetch emission tile for t+1 into the other buffer ----
#if CRF_HAVE_TDM
    if (wave == 0 && t + 1 < T_)
      crf_tdm_tile(em + ((size_t)b0 * T_ + (t + 1)) * K_,
                   (unsigned)(size_t)&ebuf[(t + 1) & 1][0][0]);
#else
    if (t + 1 < T_) {
      #pragma unroll
      for (int i = 0; i < 8; ++i)
        ebuf[(t + 1) & 1][row8][c0 + i] =
            em[((size_t)(b0 + row8) * T_ + (t + 1)) * K_ + c0 + i];
    }
#endif
    __syncthreads();                            // rowmax/ymsk visible
    // ---- split-precision exp tables: a = exp2((alpha-m)*log2e) = hi+lo ----
    {
      float mr = rowmax[row8];
      #pragma unroll
      for (int p = 0; p < 4; ++p) {
        float v0 = hw_exp2((alpha[row8][c0 + 2*p    ] - mr) * LOG2E_F);
        float v1 = hw_exp2((alpha[row8][c0 + 2*p + 1] - mr) * LOG2E_F);
        _Float16 h0 = (_Float16)v0, h1 = (_Float16)v1;
        v2h hh; hh[0] = h0; hh[1] = h1;
        v2h ll; ll[0] = (_Float16)(v0 - (float)h0);
                ll[1] = (_Float16)(v1 - (float)h1);
        *(v2h*)&ahi[row8][c0 + 2*p] = hh;
        *(v2h*)&alo[row8][c0 + 2*p] = ll;
      }
    }
    // ---- ensure this step's TDM tile has landed (in-order per wave) ----
#if CRF_HAVE_TDM
    if (wave == 0) {
      if (t < T_ - 1) __builtin_amdgcn_s_wait_tensorcnt(1);
      else            __builtin_amdgcn_s_wait_tensorcnt(0);
    }
#endif
    __syncthreads();                            // tables + emission tile ready
    // ---- A fragments (16x32 f16 layout) from LDS ----
    v16h a0h, a0l, a1h, a1l;
    {
      const int m = lane & 15;
      #pragma unroll
      for (int e = 0; e < 16; ++e) {
        int ka = e + 2 * rofs;  // K index = e + 16*(lane/16), matches B
        a0h[e] = ahi[m][ka];       a0l[e] = alo[m][ka];
        a1h[e] = ahi[m][ka + 32];  a1l[e] = alo[m][ka + 32];
      }
    }
    // ---- compensated GEMM: acc = a_hi*E_hi + a_lo*E_hi + a_hi*E_lo ----
    v8f acc = {};
    acc = __builtin_amdgcn_wmma_f32_16x16x32_f16(false, a0h, false, bh0, (short)0, acc, false, false);
    acc = __builtin_amdgcn_wmma_f32_16x16x32_f16(false, a1h, false, bh1, (short)0, acc, false, false);
    acc = __builtin_amdgcn_wmma_f32_16x16x32_f16(false, a0l, false, bh0, (short)0, acc, false, false);
    acc = __builtin_amdgcn_wmma_f32_16x16x32_f16(false, a1l, false, bh1, (short)0, acc, false, false);
    acc = __builtin_amdgcn_wmma_f32_16x16x32_f16(false, a0h, false, bl0, (short)0, acc, false, false);
    acc = __builtin_amdgcn_wmma_f32_16x16x32_f16(false, a1h, false, bl1, (short)0, acc, false, false);
    // ---- branchless masked update: alpha = old + (new-old)*mask ----
    float anew[8];
    #pragma unroll
    for (int r = 0; r < 8; ++r) {
      int m = r + rofs;
      float emv = ebuf[t & 1][m][jcol];
      float nv  = rowmax[m] + hw_log2(acc[r]) * LN2_F + emv;
      float old = alpha[m][jcol];
      anew[r] = old + (nv - old) * ymsk[m];
    }
    #pragma unroll
    for (int r = 0; r < 8; ++r) alpha[r + rofs][jcol] = anew[r];
    // (disjoint rows/cols per lane; next-iter barrier orders vs. readers)
  }

  __syncthreads();
  // ---- log_z[b] = logsumexp(alpha[row, :]) ----
  {
    float mx = alpha[row8][c0];
    #pragma unroll
    for (int c = 1; c < 8; ++c) mx = fmaxf(mx, alpha[row8][c0 + c]);
    mx = fmaxf(mx, __shfl_xor(mx, 1, 32));
    mx = fmaxf(mx, __shfl_xor(mx, 2, 32));
    mx = fmaxf(mx, __shfl_xor(mx, 4, 32));
    float s = 0.f;
    #pragma unroll
    for (int c = 0; c < 8; ++c)
      s += hw_exp2((alpha[row8][c0 + c] - mx) * LOG2E_F);
    s += __shfl_xor(s, 1, 32);
    s += __shfl_xor(s, 2, 32);
    s += __shfl_xor(s, 4, 32);
    if (sub8 == 0) logz[b0 + row8] = mx + hw_log2(s) * LN2_F;
  }
}

// ---------------------------------------------------------------------------
// Kernel 3: loss = -mean(e_scores + t_scores - log_z)
// ---------------------------------------------------------------------------
__global__ void crf_finish(const float* __restrict__ scores,
                           const float* __restrict__ logz,
                           float* __restrict__ out) {
  int tid = threadIdx.x;
  float v = scores[tid] - logz[tid];
  #pragma unroll
  for (int off = 16; off; off >>= 1) v += __shfl_xor(v, off, 32);
  __shared__ float part[8];
  if ((tid & 31) == 0) part[tid >> 5] = v;
  __syncthreads();
  if (tid == 0) {
    float tot = 0.f;
    #pragma unroll
    for (int w = 0; w < 8; ++w) tot += part[w];
    out[0] = -tot / (float)B_;
  }
}

extern "C" void kernel_launch(void* const* d_in, const int* in_sizes, int n_in,
                              void* d_out, int out_size, void* d_ws, size_t ws_size,
                              hipStream_t stream) {
  (void)in_sizes; (void)n_in; (void)out_size; (void)ws_size;
  const int*   y     = (const int*)d_in[0];
  const float* em    = (const float*)d_in[1];
  const float* trans = (const float*)d_in[2];
  float* out = (float*)d_out;

  float*    scores = (float*)d_ws;              // [256]
  float*    logzv  = scores + B_;               // [256]
  _Float16* Ehi    = (_Float16*)(logzv + B_);   // [4096]
  _Float16* Elo    = Ehi + K_ * K_;             // [4096]

  crf_prep_exp<<<(K_ * K_ + 255) / 256, 256, 0, stream>>>(trans, Ehi, Elo);
  crf_scores<<<B_, 256, 0, stream>>>(y, em, trans, scores);
  crf_scan<<<B_ / 16, 128, 0, stream>>>(y, em, Ehi, Elo, logzv);
  crf_finish<<<1, B_, 0, stream>>>(scores, logzv, out);
}
